// GNNEncoder_4750233830188
// MI455X (gfx1250) — matrix operand order repaired and checked by reading, not compiled, not run on hardware
//
#include <hip/hip_runtime.h>

// ---------------------------------------------------------------------------
// GCN (3-layer, D=128) for MI455X / gfx1250, wave32, WMMA bf16 matrix cores.
// Fusions: bias+ReLU folded into next GEMM's LDS staging; self-loop term
// replaces the aggregation-buffer zeroing; final bias folded into pooling.
// ---------------------------------------------------------------------------

typedef __attribute__((ext_vector_type(16))) __bf16 v16bf;
typedef __attribute__((ext_vector_type(8)))  __bf16 v8bf;
typedef __attribute__((ext_vector_type(8)))  float  v8f;

#define D 128        // feature dim for all layers
#define NG 64        // number of graphs

// ---------------------------------------------------------------- utilities
__global__ __launch_bounds__(256) void zero_f32(float* __restrict__ p, int n) {
    int i = blockIdx.x * blockDim.x + threadIdx.x;
    if (i < n) p[i] = 0.0f;
}

// deg[dst] += 1 over the E real edges (self loop handled in make_dinv)
__global__ __launch_bounds__(256) void deg_count(const int* __restrict__ ei,
                                                 float* __restrict__ deg, int E) {
    int e = blockIdx.x * blockDim.x + threadIdx.x;
    if (e < E) atomicAdd(&deg[ei[E + e]], 1.0f);
}

__global__ __launch_bounds__(256) void make_dinv(const float* __restrict__ deg,
                                                 float* __restrict__ dinv, int N) {
    int v = blockIdx.x * blockDim.x + threadIdx.x;
    if (v < N) dinv[v] = rsqrtf(deg[v] + 1.0f);   // +1 = self loop
}

// -------------------------------------------------- W -> bf16 B-fragment pack
// Fragment storage order: [ntile(8)][kstep(4)][lane(32)][elem(16)]
// B 32x16 fragment, contiguous k order matching the A staging: lanes 0-15 hold K=0..15 (N=lane),
// lanes 16-31 hold K=16..31 (N=lane-16); element e is contiguous in K.
__global__ __launch_bounds__(256) void wprep(const float* __restrict__ W,
                                             unsigned short* __restrict__ frag) {
    int t = blockIdx.x * blockDim.x + threadIdx.x;   // 0..16383
    int e    = t & 15;
    int lane = (t >> 4) & 31;
    int ks   = (t >> 9) & 3;
    int tile = (t >> 11) & 7;
    int K = ks * 32 + ((lane >= 16) ? 16 : 0) + e;
    int N = tile * 16 + (lane & 15);
    __bf16 v = (__bf16)W[K * D + N];
    ((__bf16*)frag)[t] = v;
}

// -------------------------------------------------------------- WMMA GEMM
// out[N,128] = act(in + bias) @ W   (W pre-packed as bf16 fragments)
// block: 256 thr = 8 waves; block handles 64 rows (4 x 16-row subtiles);
// wave w owns n-tile w and keeps its 4 B fragments in registers.
__global__ __launch_bounds__(256) void gemm_wmma(const float* __restrict__ in,
                                                 const unsigned short* __restrict__ wfrag,
                                                 const float* __restrict__ bias, int relu,
                                                 float* __restrict__ out, int nrows) {
    __shared__ __bf16 lds[64 * D];           // 16 KB A-tile in bf16
    const int row0 = blockIdx.x * 64;
    int valid = nrows - row0;
    if (valid <= 0) return;
    if (valid > 64) valid = 64;
    const int t = threadIdx.x;

    // ---- stage 64x128 fp32 -> bf16 into LDS, fusing bias (+ReLU) ----
    {
        const int srow = t >> 2;             // 0..63
        const int scol = (t & 3) * 32;       // 0,32,64,96
        if (srow < valid) {
            const float* src = in + (size_t)(row0 + srow) * D + scol;
            __bf16* dst = &lds[srow * D + scol];
#pragma unroll
            for (int c = 0; c < 32; c += 8) {
                float4 a = *(const float4*)(src + c);
                float4 b = *(const float4*)(src + c + 4);
                if (bias) {
                    float4 ba = *(const float4*)(bias + scol + c);
                    float4 bb = *(const float4*)(bias + scol + c + 4);
                    a.x += ba.x; a.y += ba.y; a.z += ba.z; a.w += ba.w;
                    b.x += bb.x; b.y += bb.y; b.z += bb.z; b.w += bb.w;
                }
                if (relu) {
                    a.x = fmaxf(a.x, 0.f); a.y = fmaxf(a.y, 0.f);
                    a.z = fmaxf(a.z, 0.f); a.w = fmaxf(a.w, 0.f);
                    b.x = fmaxf(b.x, 0.f); b.y = fmaxf(b.y, 0.f);
                    b.z = fmaxf(b.z, 0.f); b.w = fmaxf(b.w, 0.f);
                }
                v8bf v;
                v[0] = (__bf16)a.x; v[1] = (__bf16)a.y; v[2] = (__bf16)a.z; v[3] = (__bf16)a.w;
                v[4] = (__bf16)b.x; v[5] = (__bf16)b.y; v[6] = (__bf16)b.z; v[7] = (__bf16)b.w;
                *(v8bf*)(dst + c) = v;
            }
        }
    }
    __syncthreads();

    const int wave = t >> 5;                 // n-tile index 0..7
    const int lane = t & 31;
    const int hi   = (lane >= 16) ? 1 : 0;
    const int m    = lane & 15;
    const __bf16* wf = (const __bf16*)wfrag;

    // ---- B fragments held in registers across all 4 M-subtiles ----
    v16bf Bf[4];
#pragma unroll
    for (int ks = 0; ks < 4; ++ks)
        Bf[ks] = *(const v16bf*)(wf + (((wave * 4 + ks) * 32 + lane) << 4));

#pragma unroll
    for (int mt = 0; mt < 4; ++mt) {
        if (mt * 16 < valid) {               // uniform per block; EXEC stays all-1s
            v8f acc = {};
#pragma unroll
            for (int ks = 0; ks < 4; ++ks) {
                // A fragment: contiguous k order, same as the B pack:
                //   K(e,lane) = ks*32 + e + (lane>=16 ? 16 : 0)
                const __bf16* ar = &lds[(mt * 16 + m) * D + ks * 32 + hi * 16];
                v8bf a0 = *(const v8bf*)(ar);
                v8bf a1 = *(const v8bf*)(ar + 8);
                v16bf A;
#pragma unroll
                for (int i = 0; i < 8; ++i) { A[i] = a0[i]; A[i + 8] = a1[i]; }
                acc = __builtin_amdgcn_wmma_f32_16x16x32_bf16(
                          false, A, false, Bf[ks], (short)0, acc, false, false);
            }
            // C/D layout: VGPR r, lanes 0-15 -> M=r, lanes 16-31 -> M=r+8
            float* o = out + ((size_t)row0 + mt * 16 + hi * 8) * D + wave * 16 + m;
#pragma unroll
            for (int r = 0; r < 8; ++r) o[(size_t)r * D] = acc[r];
        }
    }
}

// ------------------------------------------------- self-loop initialization
// agg[v][:] = t[v][:] * dinv[v]^2   (replaces zeroing; scatter adds on top)
__global__ __launch_bounds__(256) void selfloop_init(float* __restrict__ agg,
                                                     const float* __restrict__ t,
                                                     const float* __restrict__ dinv,
                                                     int n128) {
    int i = blockIdx.x * blockDim.x + threadIdx.x;
    if (i >= n128) return;
    float di = dinv[i >> 7];
    agg[i] = t[i] * di * di;
}

// ---------------------------------------------------------- edge scatter-add
// 1 wave per edge, 4 dims per lane: agg[dst] += t[src] * dinv[src]*dinv[dst]
__global__ __launch_bounds__(256) void edge_scatter(const float* __restrict__ t,
                                                    const int* __restrict__ ei,
                                                    const float* __restrict__ dinv,
                                                    float* __restrict__ agg, int E) {
    long long tid = (long long)blockIdx.x * blockDim.x + threadIdx.x;
    int e = (int)(tid >> 5);
    if (e >= E) return;
    int lane = (int)tid & 31;
    int s = ei[e];
    int d = ei[E + e];
    float nrm = dinv[s] * dinv[d];
    const float4 v = *(const float4*)(t + (size_t)s * D + lane * 4);
    float* o = agg + (size_t)d * D + lane * 4;
    atomicAdd(o + 0, v.x * nrm);
    atomicAdd(o + 1, v.y * nrm);
    atomicAdd(o + 2, v.z * nrm);
    atomicAdd(o + 3, v.w * nrm);
}

// -------------------------------------------------------------- mean pooling
__global__ __launch_bounds__(256) void pool_sum(const float* __restrict__ h,
                                                const int* __restrict__ batch,
                                                float* __restrict__ pool,
                                                float* __restrict__ cnt, int N) {
    long long tid = (long long)blockIdx.x * blockDim.x + threadIdx.x;
    int v = (int)(tid >> 5);
    if (v >= N) return;
    int lane = (int)tid & 31;
    int g = batch[v];
    const float4 x = *(const float4*)(h + (size_t)v * D + lane * 4);
    float* o = pool + g * D + lane * 4;
    atomicAdd(o + 0, x.x);
    atomicAdd(o + 1, x.y);
    atomicAdd(o + 2, x.z);
    atomicAdd(o + 3, x.w);
    if (lane == 0) atomicAdd(&cnt[g], 1.0f);
}

// out = pool/cnt + b3   (mean(h + b) == mean(h) + b, so b3 is folded here)
__global__ __launch_bounds__(256) void pool_finalize(const float* __restrict__ pool,
                                                     const float* __restrict__ cnt,
                                                     const float* __restrict__ b3,
                                                     float* __restrict__ out) {
    int i = blockIdx.x * blockDim.x + threadIdx.x;
    if (i >= NG * D) return;
    out[i] = pool[i] / fmaxf(cnt[i >> 7], 1.0f) + b3[i & (D - 1)];
}

// ---------------------------------------------------------------------------
extern "C" void kernel_launch(void* const* d_in, const int* in_sizes, int n_in,
                              void* d_out, int out_size, void* d_ws, size_t ws_size,
                              hipStream_t stream) {
    const float* x     = (const float*)d_in[0];
    const int*   ei    = (const int*)d_in[1];     // [2][E]: src then dst
    const int*   batch = (const int*)d_in[2];
    const float* W1 = (const float*)d_in[3];
    const float* b1 = (const float*)d_in[4];
    const float* W2 = (const float*)d_in[5];
    const float* b2 = (const float*)d_in[6];
    const float* W3 = (const float*)d_in[7];
    const float* b3 = (const float*)d_in[8];
    float* out = (float*)d_out;

    const int N = in_sizes[0] / D;      // 50000
    const int E = in_sizes[1] / 2;      // 600000

    // ---- workspace carve-out (256B aligned slices) ----
    char* p = (char*)d_ws;
    auto take = [&](size_t bytes) -> char* {
        char* r = p;
        p += (bytes + 255) & ~(size_t)255;
        return r;
    };
    float* bufA = (float*)take((size_t)N * D * sizeof(float)); // GEMM output t
    float* bufB = (float*)take((size_t)N * D * sizeof(float)); // aggregation / layer io
    float* deg  = (float*)take((size_t)N * sizeof(float));
    float* dinv = (float*)take((size_t)N * sizeof(float));
    float* pool = (float*)take((size_t)NG * D * sizeof(float));
    float* cnt  = (float*)take((size_t)NG * sizeof(float));
    unsigned short* wf1 = (unsigned short*)take(8 * 4 * 32 * 16 * sizeof(unsigned short));
    unsigned short* wf2 = (unsigned short*)take(8 * 4 * 32 * 16 * sizeof(unsigned short));
    unsigned short* wf3 = (unsigned short*)take(8 * 4 * 32 * 16 * sizeof(unsigned short));

    auto cdiv = [](long long a, long long b) -> int { return (int)((a + b - 1) / b); };
    const int NB256_N  = cdiv(N, 256);
    const int NB256_ND = cdiv((long long)N * D, 256);
    const int NB_EDGE  = cdiv((long long)E * 32, 256);
    const int NB_POOL  = cdiv((long long)N * 32, 256);
    const int NB_GEMM  = cdiv(N, 64);

    // ---- normalization: deg -> dinv ----
    zero_f32 <<<NB256_N, 256, 0, stream>>>(deg, N);
    deg_count<<<cdiv(E, 256), 256, 0, stream>>>(ei, deg, E);
    make_dinv<<<NB256_N, 256, 0, stream>>>(deg, dinv, N);

    // ---- pack weights into WMMA bf16 fragment order ----
    wprep<<<64, 256, 0, stream>>>(W1, wf1);
    wprep<<<64, 256, 0, stream>>>(W2, wf2);
    wprep<<<64, 256, 0, stream>>>(W3, wf3);

    // ---- layer 1: t = x @ W1 ; agg = t*dinv^2 + scatter ----
    gemm_wmma    <<<NB_GEMM, 256, 0, stream>>>(x, wf1, nullptr, 0, bufA, N);
    selfloop_init<<<NB256_ND, 256, 0, stream>>>(bufB, bufA, dinv, N * D);
    edge_scatter <<<NB_EDGE, 256, 0, stream>>>(bufA, ei, dinv, bufB, E);

    // ---- layer 2: t = relu(agg1 + b1) @ W2 ; aggregate ----
    gemm_wmma    <<<NB_GEMM, 256, 0, stream>>>(bufB, wf2, b1, 1, bufA, N);
    selfloop_init<<<NB256_ND, 256, 0, stream>>>(bufB, bufA, dinv, N * D);
    edge_scatter <<<NB_EDGE, 256, 0, stream>>>(bufA, ei, dinv, bufB, E);

    // ---- layer 3: t = relu(agg2 + b2) @ W3 ; aggregate (b3 folded in pool) ----
    gemm_wmma    <<<NB_GEMM, 256, 0, stream>>>(bufB, wf3, b2, 1, bufA, N);
    selfloop_init<<<NB256_ND, 256, 0, stream>>>(bufB, bufA, dinv, N * D);
    edge_scatter <<<NB_EDGE, 256, 0, stream>>>(bufA, ei, dinv, bufB, E);

    // ---- global mean pool (+ b3) ----
    zero_f32     <<<cdiv(NG * D, 256), 256, 0, stream>>>(pool, NG * D);
    zero_f32     <<<1, 256, 0, stream>>>(cnt, NG);
    pool_sum     <<<NB_POOL, 256, 0, stream>>>(bufB, batch, pool, cnt, N);
    pool_finalize<<<cdiv(NG * D, 256), 256, 0, stream>>>(pool, cnt, b3, out);
}
